// FlaxGPTJAttention_45492293599389
// MI455X (gfx1250) — compile-verified, matrix operand order checked by reading, not run on hardware
//
#include <hip/hip_runtime.h>
#include <hip/hip_bf16.h>

typedef __attribute__((ext_vector_type(16))) __bf16       bf16x16;
typedef __attribute__((ext_vector_type(8)))  float        f32x8;
typedef __attribute__((ext_vector_type(4)))  unsigned int u32x4;
typedef __attribute__((ext_vector_type(4)))  int          i32x4;

#define E  4096
#define H  16
#define D  256
#define RD 64
#define S  2048
#define BB 2
#define NT (BB * S)   // 4096 tokens

__device__ __forceinline__ f32x8 zero8() {
    f32x8 z = {0.f, 0.f, 0.f, 0.f, 0.f, 0.f, 0.f, 0.f};
    return z;
}

// Load a 16x32 bf16 A/B fragment (row-major source, leading dim `ld` elems).
// ISA layout: lane<16 -> row=lane,  K = 0..7 and 16..23
//             lane>=16 -> row=lane-16, K = 8..15 and 24..31
__device__ __forceinline__ bf16x16 load_frag16(const __bf16* p, int ld) {
    const int lane = threadIdx.x & 31;
    p += (lane & 15) * ld + ((lane >> 4) << 3);
    union { u32x4 q[2]; bf16x16 v; } u;
    u.q[0] = *(const u32x4*)p;
    u.q[1] = *(const u32x4*)(p + 16);
    return u.v;
}

__device__ __forceinline__ f32x8 wmma_bf16(bf16x16 a, bf16x16 b, f32x8 c) {
    return __builtin_amdgcn_wmma_f32_16x16x32_bf16(
        /*neg_a=*/false, a, /*neg_b=*/false, b,
        /*c_mod=*/(short)0, c, /*reuse_a=*/false, /*reuse_b=*/false);
}

// ---- gfx1250 async global->LDS copy (16 bytes per lane) -------------------
__device__ __forceinline__ void async_cp16(const __bf16* g, __bf16* l) {
#if __has_builtin(__builtin_amdgcn_global_load_async_to_lds_b128)
    __builtin_amdgcn_global_load_async_to_lds_b128(
        (__attribute__((address_space(1))) i32x4*)(g),
        (__attribute__((address_space(3))) i32x4*)(l), 0, 0);
#else
    unsigned int loff =
        (unsigned int)(size_t)(__attribute__((address_space(3))) void*)l;
    asm volatile("global_load_async_to_lds_b128 %0, %1, off"
                 :: "v"(loff), "v"(g) : "memory");
#endif
}

__device__ __forceinline__ void wait_async0() {
#if __has_builtin(__builtin_amdgcn_s_wait_asynccnt)
    __builtin_amdgcn_s_wait_asynccnt(0);
#else
    asm volatile("s_wait_asynccnt 0" ::: "memory");
#endif
}

// ---------------------------------------------------------------------------
// Tiled WMMA GEMM: C[M=4096][N=4096] = A[M][K=4096] * Bw[K][N]  (Bw is fp32)
// OUT_MODE 0: fp32, plain [m][n]
// OUT_MODE 2: bf16, V-transposed  [B,H,D,S]   (m->b,s ; n->h,d)
// OUT_MODE 3: bf16, head-permuted [B,H,S,D]
// ---------------------------------------------------------------------------
template <bool A_F32, int OUT_MODE>
__global__ __launch_bounds__(256) void gemm_kernel(const void* __restrict__ Ap,
                                                   const float* __restrict__ Bw,
                                                   void* __restrict__ Cp,
                                                   float scale) {
    __shared__ __bf16 lA[128][40];   // padded to 40 to reduce bank conflicts
    __shared__ __bf16 lB[128][40];   // lB[n][k] (transposed while loading)

    const int tid  = threadIdx.x;
    const int lane = tid & 31;
    const int wave = tid >> 5;
    const int m0 = blockIdx.y * 128;
    const int n0 = blockIdx.x * 128;
    const int wm = wave >> 1;        // 0..3 -> 32 rows each
    const int wn = wave & 1;         // 0..1 -> 64 cols each

    f32x8 acc[2][4];
#pragma unroll
    for (int i = 0; i < 2; i++)
#pragma unroll
        for (int j = 0; j < 4; j++) acc[i][j] = zero8();

    const int arow = tid >> 1;             // 0..127
    const int aseg = (tid & 1) * 16;       // 0 / 16
    const int bk   = tid >> 3;             // 0..31
    const int bnsg = (tid & 7) * 16;       // 0..112

    for (int kk = 0; kk < E; kk += 32) {
        // ---- stage A tile (128 x 32) into LDS, converting if fp32 ----
        if (A_F32) {
            const float* ap = (const float*)Ap + (size_t)(m0 + arow) * E + kk + aseg;
#pragma unroll
            for (int i = 0; i < 16; i += 4) {
                float4 f = *(const float4*)(ap + i);
                lA[arow][aseg + i + 0] = (__bf16)f.x;
                lA[arow][aseg + i + 1] = (__bf16)f.y;
                lA[arow][aseg + i + 2] = (__bf16)f.z;
                lA[arow][aseg + i + 3] = (__bf16)f.w;
            }
        } else {
            const __bf16* ap = (const __bf16*)Ap + (size_t)(m0 + arow) * E + kk + aseg;
            *(u32x4*)&lA[arow][aseg]     = *(const u32x4*)ap;
            *(u32x4*)&lA[arow][aseg + 8] = *(const u32x4*)(ap + 8);
        }
        // ---- stage B tile (32 x 128 fp32) transposed -> lB[n][k] bf16 ----
        {
            const float* bp = Bw + (size_t)(kk + bk) * E + n0 + bnsg;
#pragma unroll
            for (int i = 0; i < 16; i += 4) {
                float4 f = *(const float4*)(bp + i);
                lB[bnsg + i + 0][bk] = (__bf16)f.x;
                lB[bnsg + i + 1][bk] = (__bf16)f.y;
                lB[bnsg + i + 2][bk] = (__bf16)f.z;
                lB[bnsg + i + 3][bk] = (__bf16)f.w;
            }
        }
        __syncthreads();

        bf16x16 afr0 = load_frag16(&lA[wm * 32][0], 40);
        bf16x16 afr1 = load_frag16(&lA[wm * 32 + 16][0], 40);
#pragma unroll
        for (int t = 0; t < 4; t++) {
            bf16x16 bfr = load_frag16(&lB[wn * 64 + t * 16][0], 40);
            acc[0][t] = wmma_bf16(afr0, bfr, acc[0][t]);
            acc[1][t] = wmma_bf16(afr1, bfr, acc[1][t]);
        }
        __syncthreads();
    }

    // ---- epilogue (C layout: lane = col, vgpr r + half = row) ----
#pragma unroll
    for (int sub = 0; sub < 2; sub++)
#pragma unroll
        for (int t = 0; t < 4; t++)
#pragma unroll
            for (int r = 0; r < 8; r++) {
                const int mAbs = m0 + wm * 32 + sub * 16 + ((lane >> 4) << 3) + r;
                const int nAbs = n0 + wn * 64 + t * 16 + (lane & 15);
                const float v = acc[sub][t][r] * scale;
                if (OUT_MODE == 0) {
                    ((float*)Cp)[(size_t)mAbs * E + nAbs] = v;
                } else {
                    const int b = mAbs >> 11, s = mAbs & (S - 1);
                    const int h = nAbs >> 8,  d = nAbs & (D - 1);
                    if (OUT_MODE == 3)
                        ((__bf16*)Cp)[((size_t)((b * H + h) * S + s)) * D + d] = (__bf16)v;
                    else
                        ((__bf16*)Cp)[((size_t)((b * H + h) * D + d)) * S + s] = (__bf16)v;
                }
            }
}

// ---------------------------------------------------------------------------
// In-place interleaved RoPE on first RD dims of each head, layout [B,H,S,D].
// One thread per (b,h,s,pair). Rotation commutes with the folded 1/sqrt(D).
// ---------------------------------------------------------------------------
__global__ void rope_kernel(__bf16* __restrict__ buf, const int* __restrict__ pos) {
    const int idx = blockIdx.x * blockDim.x + threadIdx.x;
    if (idx >= BB * H * S * (RD / 2)) return;
    const int j = idx & 31;                // rotary pair index 0..31
    const int s = (idx >> 5) & (S - 1);
    const int h = (idx >> 16) & (H - 1);
    const int b = idx >> 20;

    __bf16* p = buf + ((size_t)((b * H + h) * S + s)) * D + 2 * j;
    const float t0 = (float)p[0];
    const float t1 = (float)p[1];
    const float invf = __powf(10000.f, -(2.f * (float)j) / (float)RD);
    const float ang = (float)pos[b * S + s] * invf;
    float sn, cs;
    __sincosf(ang, &sn, &cs);
    p[0] = (__bf16)(t0 * cs - t1 * sn);
    p[1] = (__bf16)(t1 * cs + t0 * sn);
}

// ---------------------------------------------------------------------------
// Flash attention: one block per (64-query tile, head, batch); 8 waves =
// 4 query-subtiles x 2 dhead-halves. Key blocks of 32.
// K (32x256) and V^T (256x32) tiles are staged in LDS once per block via
// gfx1250 async global->LDS copies (shared by all 8 waves), fp32 online
// softmax, P restaged via wave-private LDS (C-layout -> A-fragment layout).
// Q scale (1/16) already folded into Q.
// ---------------------------------------------------------------------------
__global__ __launch_bounds__(256) void attn_kernel(const __bf16* __restrict__ Qp,
                                                   const __bf16* __restrict__ Kp,
                                                   const __bf16* __restrict__ Vt,
                                                   const int* __restrict__ amask,
                                                   __bf16* __restrict__ AO) {
    __shared__ __bf16 lK[32][264];     // 32 keys x 256 dims (pad 8)   ~16.5 KB
    __shared__ __bf16 lV[256][40];     // 256 dims x 32 keys (pad 8)   ~20 KB
    __shared__ __bf16 lP[8][16][40];   // per-wave P staging           ~10 KB

    const int tid  = threadIdx.x;
    const int lane = tid & 31;
    const int wave = tid >> 5;
    const int qtile = blockIdx.x, h = blockIdx.y, b = blockIdx.z;
    const int qsub  = wave >> 1;
    const int dbase = (wave & 1) * 128;
    const int row0  = qtile * 64 + qsub * 16;

    const __bf16* Qh = Qp + ((size_t)(b * H + h) * S) * D;
    const __bf16* Kh = Kp + ((size_t)(b * H + h) * S) * D;
    const __bf16* Vh = Vt + ((size_t)(b * H + h) * D) * S;

    // Q fragments for this wave's 16 rows, full D=256 (8 k-blocks of 32)
    bf16x16 aQ[8];
#pragma unroll
    for (int i = 0; i < 8; i++)
        aQ[i] = load_frag16(Qh + (size_t)row0 * D + i * 32, D);

    f32x8 o[8];
#pragma unroll
    for (int i = 0; i < 8; i++) o[i] = zero8();
    float m[8], l[8];
#pragma unroll
    for (int r = 0; r < 8; r++) { m[r] = -1e30f; l[r] = 0.f; }

    const int jend = qtile * 64 + 64;   // causal upper bound for this tile
    for (int j0 = 0; j0 < jend; j0 += 32) {
        __syncthreads();   // previous iteration's LDS reads are complete
        // ---- async stage K tile: 32 rows x 512B = 1024 x 16B chunks ----
#pragma unroll
        for (int i = 0; i < 4; i++) {
            const int c = tid + 256 * i;
            const int kr = c >> 5, kc = c & 31;
            async_cp16(Kh + (size_t)(j0 + kr) * D + kc * 8, &lK[kr][kc * 8]);
        }
        // ---- async stage V tile: 256 rows x 64B = 1024 x 16B chunks ----
#pragma unroll
        for (int i = 0; i < 4; i++) {
            const int c = tid + 256 * i;
            const int vr = c >> 2, vc = c & 3;
            async_cp16(Vh + (size_t)vr * S + j0 + vc * 8, &lV[vr][vc * 8]);
        }
        wait_async0();     // own async ops done
        __syncthreads();   // everyone's async ops done

        // ---- scores 16x32 = Q(16x256) . K^T : two 16-col WMMA tiles ----
        f32x8 sc[2];
        sc[0] = zero8();
        sc[1] = zero8();
#pragma unroll
        for (int t = 0; t < 2; t++)
#pragma unroll
            for (int i = 0; i < 8; i++) {
                bf16x16 bk = load_frag16(&lK[t * 16][i * 32], 264);
                sc[t] = wmma_bf16(aQ[i], bk, sc[t]);
            }
        // ---- causal + padding mask ----
#pragma unroll
        for (int t = 0; t < 2; t++) {
            const int kAbs = j0 + t * 16 + (lane & 15);
            const bool kok = amask[b * S + kAbs] > 0;
#pragma unroll
            for (int r = 0; r < 8; r++) {
                const int qAbs = row0 + ((lane >> 4) << 3) + r;
                float v = sc[t][r];
                if (kAbs > qAbs || !kok) v = -1e30f;
                sc[t][r] = v;
            }
        }
        // ---- online softmax (row reductions across the 16 col-lanes) ----
#pragma unroll
        for (int r = 0; r < 8; r++) {
            float rm = fmaxf(sc[0][r], sc[1][r]);
#pragma unroll
            for (int msk = 1; msk < 16; msk <<= 1)
                rm = fmaxf(rm, __shfl_xor(rm, msk, 32));
            const float mnew  = fmaxf(m[r], rm);
            const float alpha = __expf(m[r] - mnew);
            const float p0 = __expf(sc[0][r] - mnew);
            const float p1 = __expf(sc[1][r] - mnew);
            float rs = p0 + p1;
#pragma unroll
            for (int msk = 1; msk < 16; msk <<= 1)
                rs += __shfl_xor(rs, msk, 32);
            l[r] = l[r] * alpha + rs;
            m[r] = mnew;
#pragma unroll
            for (int i = 0; i < 8; i++) o[i][r] *= alpha;
            const int Mr = ((lane >> 4) << 3) + r;
            lP[wave][Mr][lane & 15]        = (__bf16)p0;   // wave-private, in-order LDS
            lP[wave][Mr][16 + (lane & 15)] = (__bf16)p1;
        }
        // ---- O += P(16x32) . V(32x128-half) ----
        bf16x16 aP = load_frag16(&lP[wave][0][0], 40);
#pragma unroll
        for (int ns = 0; ns < 8; ns++) {
            bf16x16 bv = load_frag16(&lV[dbase + ns * 16][0], 40);
            o[ns] = wmma_bf16(aP, bv, o[ns]);
        }
    }
    // ---- epilogue: normalize, write bf16 [token][E] ----
#pragma unroll
    for (int ns = 0; ns < 8; ns++)
#pragma unroll
        for (int r = 0; r < 8; r++) {
            const int qAbs = row0 + ((lane >> 4) << 3) + r;
            const int col  = h * D + dbase + ns * 16 + (lane & 15);
            AO[(size_t)(b * S + qAbs) * E + col] = (__bf16)(o[ns][r] / l[r]);
        }
}

// ---------------------------------------------------------------------------
extern "C" void kernel_launch(void* const* d_in, const int* in_sizes, int n_in,
                              void* d_out, int out_size, void* d_ws, size_t ws_size,
                              hipStream_t stream) {
    const float* X     = (const float*)d_in[0];
    const int*   amask = (const int*)d_in[1];
    const int*   pos   = (const int*)d_in[2];
    const float* Wq    = (const float*)d_in[3];
    const float* Wk    = (const float*)d_in[4];
    const float* Wv    = (const float*)d_in[5];
    const float* Wo    = (const float*)d_in[6];

    __bf16* Qp = (__bf16*)d_ws;                 // [B,H,S,D]  32 MB
    __bf16* Kp = Qp + (size_t)NT * E;           // [B,H,S,D]  32 MB
    __bf16* Vt = Kp + (size_t)NT * E;           // [B,H,D,S]  32 MB
    __bf16* AO = Vt + (size_t)NT * E;           // [token][E] 32 MB

    const dim3 gg(E / 128, NT / 128);
    const dim3 bb(256);

    // QKV projections (1/sqrt(D)=0.0625 folded into Q)
    gemm_kernel<true, 3><<<gg, bb, 0, stream>>>((const void*)X, Wq, (void*)Qp, 0.0625f);
    gemm_kernel<true, 3><<<gg, bb, 0, stream>>>((const void*)X, Wk, (void*)Kp, 1.0f);
    gemm_kernel<true, 2><<<gg, bb, 0, stream>>>((const void*)X, Wv, (void*)Vt, 1.0f);

    const int nrope = BB * H * S * (RD / 2);
    rope_kernel<<<nrope / 256, 256, 0, stream>>>(Qp, pos);
    rope_kernel<<<nrope / 256, 256, 0, stream>>>(Kp, pos);

    attn_kernel<<<dim3(S / 64, H, BB), bb, 0, stream>>>(Qp, Kp, Vt, amask, AO);

    // output projection -> fp32 d_out
    gemm_kernel<false, 0><<<gg, bb, 0, stream>>>((const void*)AO, Wo, d_out, 1.0f);
}
